// DeformableConv2d_43293270343950
// MI455X (gfx1250) — matrix operand order repaired and checked by reading, not compiled, not run on hardware
//
#include <hip/hip_runtime.h>

// ---------------------------------------------------------------------------
// Fused deformable-conv-v2 for MI455X (gfx1250, wave32, WMMA).
//  Stage 0 (pack kernel): pre-convert all weights into bf16 WMMA A-fragments
//     in d_ws, in exact wave32 register layout (32 lanes x 32B per fragment).
//  Stage 1 (fused kernel, per 64-pixel tile):
//     Phase 1: offset/mask 3x3 conv as WMMA GEMM [32 x 576] x [576 x 64],
//              im2col tile staged in LDS as bf16 (double-buffered chunks,
//              branchless clamped loads, register-incremental k decode).
//     Phase 2: bilinear corner setup per (pixel, tap); mask & validity folded
//              into 4 corner weights + 4 clipped flat indices in LDS.
//     Phase 3: deformable GEMM out[64,64] = W[64,576] x samp[576,64] on
//              v_wmma_f32_16x16x32_bf16, K chunked by 96, double-buffered.
// ---------------------------------------------------------------------------

#define BATCH  4
#define CIN    64
#define HH     128
#define WW     128
#define OUTC   64
#define K2N    9
#define KDIM   576      // CIN * K2N
#define HWSZ   16384    // HH * WW  (== 1 << 14)
#define NT     64       // output pixels per workgroup
#define KC     96       // K-chunk (3 WMMA k-steps of 32)
#define NCHUNK 6        // KDIM / KC
#define NKSTEP 18       // KDIM / 32
#define KSTR   104      // padded bf16 row stride for LDS B tile (16B-aligned)
#define QY_LUT 0x2A540  // 2 bits per q: qy(q) for q = 0..8

// d_ws layout: [0..72KB) deform A-frags (18 ksteps x 4 mtiles),
//              [72KB..108KB) conv A-frags (18 ksteps x 2 mtiles).
#define NFRAG_DEF  (NKSTEP * 4)
#define NFRAG_CONV (NKSTEP * 2)
#define NFRAG      (NFRAG_DEF + NFRAG_CONV)         // 108 fragments
#define FRAG_BYTES 1024                             // 32 lanes x 32B
#define WS_NEEDED  ((size_t)NFRAG * FRAG_BYTES)     // 110592 B

typedef __attribute__((ext_vector_type(8)))  float   v8f;
typedef __attribute__((ext_vector_type(16))) __bf16  v16bf;

union BFrag {
    v16bf        v;
    unsigned int u[8];
    uint4        q[2];
};

static __device__ __forceinline__ unsigned short f32_to_bf16(float f) {
    unsigned int u = __float_as_uint(f);
    u += 0x7FFFu + ((u >> 16) & 1u);        // round-to-nearest-even
    return (unsigned short)(u >> 16);
}
static __device__ __forceinline__ unsigned int pack2_bf16(float lo, float hi) {
    return (unsigned int)f32_to_bf16(lo) | ((unsigned int)f32_to_bf16(hi) << 16);
}

// ---------------------------------------------------------------------------
// Pack kernel: build every A-fragment once.
// A-fragment layout (16-bit A 16x32, wave32): lane 0-15 => M=lane, K-phase 0;
// lane 16-31 => M=lane-16, K-phase 8. dword t8 holds K = phase + (t8<4 ? 2*t8
// : 8+2*t8) and +1.
// ---------------------------------------------------------------------------
__global__ __launch_bounds__(128)
void pack_wfrags_kernel(const float* __restrict__ offw,
                        const float* __restrict__ modw,
                        const float* __restrict__ weight,
                        uint4* __restrict__ ws)
{
    const int gid  = blockIdx.x * blockDim.x + threadIdx.x;  // 27*128 = 3456
    const int frag = gid >> 5;
    const int lane = gid & 31;
    if (frag >= NFRAG) return;

    const int mm = lane & 15;
    const int ko = (lane < 16) ? 0 : 8;

    int kt, row;
    const float* src;
    if (frag < NFRAG_DEF) {                 // deform GEMM weights: [64 x 576]
        kt  = frag >> 2;
        row = (frag & 3) * 16 + mm;
        src = weight + (size_t)row * KDIM;
    } else {                                // conv weights: rows 0-17 offset,
        const int fc = frag - NFRAG_DEF;    // 18-26 mask, 27-31 zero pad
        kt  = fc >> 1;
        row = (fc & 1) * 16 + mm;
        src = (row < 18) ? (offw + (size_t)row * KDIM)
            : (row < 27) ? (modw + (size_t)(row - 18) * KDIM)
            : nullptr;
    }

    BFrag af;
    const int kbase = kt * 32 + ko;
    #pragma unroll
    for (int t8 = 0; t8 < 8; ++t8) {
        const int kloc = (t8 < 4) ? (2 * t8) : (8 + 2 * t8);
        if (src) {
            const float2 f2 = *(const float2*)(src + kbase + kloc);
            af.u[t8] = pack2_bf16(f2.x, f2.y);
        } else {
            af.u[t8] = 0u;
        }
    }
    uint4* dst = ws + (size_t)frag * (FRAG_BYTES / 16) + lane * 2;
    dst[0] = af.q[0];
    dst[1] = af.q[1];
}

// ---------------------------------------------------------------------------
// Main fused kernel.
// ---------------------------------------------------------------------------
template <bool PRE>
__global__ __launch_bounds__(128)
void deform_conv_fused_kernel(const float* __restrict__ x,
                              const float* __restrict__ offw,
                              const float* __restrict__ offb,
                              const float* __restrict__ modw,
                              const float* __restrict__ modb,
                              const float* __restrict__ weight,
                              const uint4* __restrict__ wfrag,
                              float* __restrict__ out)
{
    __shared__ float s_w[K2N * NT][4];                  // corner weights
    __shared__ int   s_i[K2N * NT][4];                  // corner indices
    __shared__ __align__(16) unsigned short s_bt[2][NT][KSTR];  // bf16 chunks
    __shared__ float s_conv[32][NT];                    // conv results (M x n)

    const int tid  = threadIdx.x;
    const int lane = tid & 31;
    const int wave = tid >> 5;

    const int b      = blockIdx.x >> 8;            // 256 tiles per image
    const int hwbase = (blockIdx.x & 255) * NT;
    const float* xb  = x + ((size_t)b << 20);      // b * CIN * HWSZ

    const int nn    = lane & 15;
    const int koA   = (lane < 16) ? 0 : 8;   // A frag K-phase (interleaved)
    const int koB   = (lane < 16) ? 0 : 8;   // B frag K-phase (interleaved)
    const int rbase = (lane < 16) ? 0 : 8;   // C/D row phase

    // staging geometry: per-thread constants
    const int n_st  = tid & (NT - 1);        // pixel this thread stages
    const int kkph  = (tid >> 6) << 1;       // kk phase: 0 or 2
    const int hw_st = hwbase + n_st;
    const int yb_st = (hw_st >> 7) - 1;
    const int xs_st = (hw_st & (WW - 1)) - 1;

    // incremental k -> (c, q) decode state; k = kkph + j + 4*step
    int cj[2], qj[2];

    // stages one K-chunk of the conv im2col tile into bt
    auto stage_conv = [&](unsigned short (*bt)[KSTR]) {
        int kk0 = kkph;
        for (int i = 0; i < (NT * KC) / (2 * 128); ++i) {   // 24 iterations
            float v[2];
            #pragma unroll
            for (int j = 0; j < 2; ++j) {
                const int q  = qj[j];
                const int qy = (QY_LUT >> (2 * q)) & 3;
                const int qx = q - 3 * qy;
                const int y  = yb_st + qy;
                const int xx = xs_st + qx;
                const float okf = (((unsigned)y < (unsigned)HH) &&
                                   ((unsigned)xx < (unsigned)WW)) ? 1.0f : 0.0f;
                const int yc = min(max(y, 0), HH - 1);
                const int xc = min(max(xx, 0), WW - 1);
                v[j] = xb[(cj[j] << 14) + (yc << 7) + xc] * okf;
                qj[j] += 4;
                if (qj[j] >= 9) { qj[j] -= 9; ++cj[j]; }
            }
            *(unsigned int*)&bt[n_st][kk0] = pack2_bf16(v[0], v[1]);
            kk0 += 4;
        }
    };

    // stages one K-chunk of the sampled (deform) B tile into bt
    auto stage_samp = [&](unsigned short (*bt)[KSTR]) {
        int kk0 = kkph;
        for (int i = 0; i < (NT * KC) / (2 * 128); ++i) {   // 24 iterations
            float v[2];
            #pragma unroll
            for (int j = 0; j < 2; ++j) {
                const int t = (qj[j] << 6) + n_st;
                const float* xc = xb + (cj[j] << 14);
                v[j] = s_w[t][0] * xc[s_i[t][0]]
                     + s_w[t][1] * xc[s_i[t][1]]
                     + s_w[t][2] * xc[s_i[t][2]]
                     + s_w[t][3] * xc[s_i[t][3]];
                qj[j] += 4;
                if (qj[j] >= 9) { qj[j] -= 9; ++cj[j]; }
            }
            *(unsigned int*)&bt[n_st][kk0] = pack2_bf16(v[0], v[1]);
            kk0 += 4;
        }
    };

    // ================= Phase 1: offset/mask conv via WMMA ==================
    v8f accc[2];
    #pragma unroll
    for (int mt = 0; mt < 2; ++mt)
        #pragma unroll
        for (int r = 0; r < 8; ++r) accc[mt][r] = 0.0f;

    cj[0] = 0; qj[0] = kkph;
    cj[1] = 0; qj[1] = kkph + 1;
    stage_conv(s_bt[0]);
    __syncthreads();

    for (int kc = 0; kc < NCHUNK; ++kc) {
        if (kc + 1 < NCHUNK) stage_conv(s_bt[(kc + 1) & 1]);

        const unsigned short (*bt)[KSTR] = s_bt[kc & 1];
        const int k0 = kc * KC;
        #pragma unroll
        for (int ks = 0; ks < 3; ++ks) {
            const int kb = ks * 32;
            BFrag bf;
            {
                const uint4* p = (const uint4*)
                    ((const unsigned int*)&bt[wave * 16 + nn][0]
                     + ((kb + koB) >> 1));
                bf.q[0] = p[0];
                bf.q[1] = p[2];
            }
            const int kt = (k0 + kb) >> 5;
            #pragma unroll
            for (int mt = 0; mt < 2; ++mt) {
                BFrag af;
                if (PRE) {
                    const uint4* p = wfrag
                        + (size_t)(NFRAG_DEF + kt * 2 + mt) * (FRAG_BYTES / 16)
                        + lane * 2;
                    af.q[0] = p[0];
                    af.q[1] = p[1];
                } else {
                    const int row = mt * 16 + nn;
                    const float* src =
                        (row < 18) ? (offw + (size_t)row * KDIM)
                      : (row < 27) ? (modw + (size_t)(row - 18) * KDIM)
                      : nullptr;
                    #pragma unroll
                    for (int t8 = 0; t8 < 8; ++t8) {
                        const int kloc = (t8 < 4) ? (2 * t8) : (8 + 2 * t8);
                        if (src) {
                            const float2 f2 =
                                *(const float2*)(src + kt * 32 + koA + kloc);
                            af.u[t8] = pack2_bf16(f2.x, f2.y);
                        } else af.u[t8] = 0u;
                    }
                }
                accc[mt] = __builtin_amdgcn_wmma_f32_16x16x32_bf16(
                    false, af.v, false, bf.v, (short)0, accc[mt], false, false);
            }
        }
        __syncthreads();
    }

    // park conv results in LDS (C/D layout: N = lane&15, M = r + rbase)
    #pragma unroll
    for (int mt = 0; mt < 2; ++mt)
        #pragma unroll
        for (int r = 0; r < 8; ++r)
            s_conv[mt * 16 + rbase + r][wave * 16 + nn] = accc[mt][r];
    __syncthreads();

    // ================= Phase 2: bilinear corner setup ======================
    for (int t = tid; t < K2N * NT; t += blockDim.x) {
        const int n  = t & (NT - 1);
        const int k2 = t >> 6;
        const int hw = hwbase + n;
        const int ho = hw >> 7;
        const int wo = hw & (WW - 1);

        const float sdy = s_conv[2 * k2][n]     + offb[2 * k2];
        const float sdx = s_conv[2 * k2 + 1][n] + offb[2 * k2 + 1];
        const float sm  = s_conv[18 + k2][n]    + modb[k2];

        const float m  = 2.0f / (1.0f + __expf(-sm));   // 2*sigmoid
        const int   ky = (QY_LUT >> (2 * k2)) & 3;
        const int   kx = k2 - 3 * ky;
        const float py = (float)(ho - 1 + ky) + sdy;
        const float px = (float)(wo - 1 + kx) + sdx;
        const float y0f = floorf(py), x0f = floorf(px);
        const float wy1 = py - y0f,   wx1 = px - x0f;
        const float wy0 = 1.0f - wy1, wx0 = 1.0f - wx1;
        const int y0 = (int)y0f, x0 = (int)x0f;
        const int y1 = y0 + 1,   x1 = x0 + 1;

        const float v00 = ((unsigned)y0 < HH && (unsigned)x0 < WW) ? 1.0f : 0.0f;
        const float v01 = ((unsigned)y0 < HH && (unsigned)x1 < WW) ? 1.0f : 0.0f;
        const float v10 = ((unsigned)y1 < HH && (unsigned)x0 < WW) ? 1.0f : 0.0f;
        const float v11 = ((unsigned)y1 < HH && (unsigned)x1 < WW) ? 1.0f : 0.0f;
        const int yc0 = min(max(y0, 0), HH - 1), yc1 = min(max(y1, 0), HH - 1);
        const int xc0 = min(max(x0, 0), WW - 1), xc1 = min(max(x1, 0), WW - 1);

        s_w[t][0] = wy0 * wx0 * m * v00;  s_i[t][0] = yc0 * WW + xc0;
        s_w[t][1] = wy0 * wx1 * m * v01;  s_i[t][1] = yc0 * WW + xc1;
        s_w[t][2] = wy1 * wx0 * m * v10;  s_i[t][2] = yc1 * WW + xc0;
        s_w[t][3] = wy1 * wx1 * m * v11;  s_i[t][3] = yc1 * WW + xc1;
    }
    __syncthreads();

    // ================= Phase 3: deformable GEMM via WMMA ===================
    v8f acc[4];
    #pragma unroll
    for (int mt = 0; mt < 4; ++mt)
        #pragma unroll
        for (int r = 0; r < 8; ++r) acc[mt][r] = 0.0f;

    cj[0] = 0; qj[0] = kkph;
    cj[1] = 0; qj[1] = kkph + 1;
    stage_samp(s_bt[0]);
    __syncthreads();

    for (int kc = 0; kc < NCHUNK; ++kc) {
        if (kc + 1 < NCHUNK) stage_samp(s_bt[(kc + 1) & 1]);

        const unsigned short (*bt)[KSTR] = s_bt[kc & 1];
        const int k0 = kc * KC;
        #pragma unroll
        for (int ks = 0; ks < 3; ++ks) {
            const int kb = ks * 32;
            BFrag bf;
            {
                const uint4* p = (const uint4*)
                    ((const unsigned int*)&bt[wave * 16 + nn][0]
                     + ((kb + koB) >> 1));
                bf.q[0] = p[0];
                bf.q[1] = p[2];
            }
            const int kt = (k0 + kb) >> 5;
            #pragma unroll
            for (int mt = 0; mt < 4; ++mt) {
                BFrag af;
                if (PRE) {
                    const uint4* p = wfrag
                        + (size_t)(kt * 4 + mt) * (FRAG_BYTES / 16) + lane * 2;
                    af.q[0] = p[0];
                    af.q[1] = p[1];
                } else {
                    const float* wrow =
                        weight + (size_t)(mt * 16 + nn) * KDIM + kt * 32 + koA;
                    #pragma unroll
                    for (int t8 = 0; t8 < 8; ++t8) {
                        const int kloc = (t8 < 4) ? (2 * t8) : (8 + 2 * t8);
                        const float2 f2 = *(const float2*)(wrow + kloc);
                        af.u[t8] = pack2_bf16(f2.x, f2.y);
                    }
                }
                acc[mt] = __builtin_amdgcn_wmma_f32_16x16x32_bf16(
                    false, af.v, false, bf.v, (short)0, acc[mt], false, false);
            }
        }
        __syncthreads();
    }

    // ================= store 64x64 f32 tile ================================
    const int hw = hwbase + wave * 16 + nn;
    float* ob = out + ((size_t)b << 20) + hw;
    #pragma unroll
    for (int mt = 0; mt < 4; ++mt)
        #pragma unroll
        for (int r = 0; r < 8; ++r)
            ob[(size_t)(mt * 16 + rbase + r) << 14] = acc[mt][r];
}

extern "C" void kernel_launch(void* const* d_in, const int* in_sizes, int n_in,
                              void* d_out, int out_size, void* d_ws, size_t ws_size,
                              hipStream_t stream) {
    (void)in_sizes; (void)n_in; (void)out_size;
    const float* x    = (const float*)d_in[0];
    const float* offw = (const float*)d_in[1];
    const float* offb = (const float*)d_in[2];
    const float* modw = (const float*)d_in[3];
    const float* modb = (const float*)d_in[4];
    const float* wgt  = (const float*)d_in[5];
    float* out = (float*)d_out;

    const dim3 grid(BATCH * HWSZ / NT);   // 1024 workgroups
    const dim3 block(128);                // 4 wave32s

    if (ws_size >= WS_NEEDED && d_ws != nullptr) {
        uint4* ws = (uint4*)d_ws;
        hipLaunchKernelGGL(pack_wfrags_kernel,
                           dim3((NFRAG * 32) / 128), dim3(128), 0, stream,
                           offw, modw, wgt, ws);
        hipLaunchKernelGGL((deform_conv_fused_kernel<true>),
                           grid, block, 0, stream,
                           x, offw, offb, modw, modb, wgt,
                           (const uint4*)ws, out);
    } else {
        hipLaunchKernelGGL((deform_conv_fused_kernel<false>),
                           grid, block, 0, stream,
                           x, offw, offb, modw, modb, wgt,
                           (const uint4*)nullptr, out);
    }
}
